// LocalAttentionLayer_41755672052347
// MI455X (gfx1250) — compile-verified, matrix operand order checked by reading, not run on hardware
//
#include <hip/hip_runtime.h>
#include <hip/hip_bf16.h>

typedef __attribute__((ext_vector_type(16))) __bf16 v16bf;
typedef __attribute__((ext_vector_type(8)))  float  v8f;

#define NPIX 4096
#define DIMD 256

// ---------------------------------------------------------------------------
// Weight convert: W [K x N] f32 (row-major) -> WT [N x K] bf16 (row-major)
// so WMMA B-fragments (lane = column, 16 consecutive K) load contiguously.
// ---------------------------------------------------------------------------
__global__ __launch_bounds__(256)
void wconv_kernel(const float* __restrict__ W, __bf16* __restrict__ WT, int K, int N)
{
    int idx = blockIdx.x * 256 + threadIdx.x;
    if (idx < K * N) {
        int n = idx / K;
        int k = idx - n * K;
        WT[idx] = (__bf16)W[(size_t)k * N + n];
    }
}

// ---------------------------------------------------------------------------
// Prep: recomputed active mask (act && !(i==2&&j==2)), masked x (f32 + bf16),
// memory in bf16.
// ---------------------------------------------------------------------------
__global__ __launch_bounds__(256)
void prep_kernel(const float* __restrict__ x, const float* __restrict__ mem,
                 const unsigned char* __restrict__ act,
                 float* __restrict__ actf, float* __restrict__ Xm,
                 __bf16* __restrict__ Xbf, __bf16* __restrict__ Mbf)
{
    int pix = blockIdx.x, t = threadIdx.x;
    int i = pix >> 6, j = pix & 63;
    float a = (act[pix] != 0 && !(i == 2 && j == 2)) ? 1.0f : 0.0f;
    if (t == 0) actf[pix] = a;
    size_t idx = (size_t)pix * DIMD + t;
    float xv = x[idx] * a;
    Xm[idx]  = xv;
    Xbf[idx] = (__bf16)xv;
    Mbf[idx] = (__bf16)mem[idx];
}

// ---------------------------------------------------------------------------
// bf16 WMMA GEMM: C[M x N] = A[M x K] @ B[K x N] + bias,  BT = B^T [N x K].
// Block = 8 waves; wave grid 4x2; each wave owns a 32x32 tile (4 accumulators).
// Optional f32 and/or bf16 output, optional fused ReLU.
// ---------------------------------------------------------------------------
__global__ __launch_bounds__(256)
void gemm_bf16_kernel(const __bf16* __restrict__ A, const __bf16* __restrict__ BT,
                      const float* __restrict__ bias,
                      float* __restrict__ Cf, __bf16* __restrict__ Cbf,
                      int N, int K, int relu)
{
    const int lane = threadIdx.x & 31;
    const int wave = threadIdx.x >> 5;
    const int wm = wave & 3, wn = wave >> 2;
    const int m0 = blockIdx.x * 128 + wm * 32;
    const int n0 = blockIdx.y * 64  + wn * 32;
    const int half = lane >> 4;     // K-half selector: contiguous k = 16*half + s, same for A and B
    const int l16  = lane & 15;

    const __bf16* arow0 = A  + (size_t)(m0 + l16)      * K + half * 16;
    const __bf16* arow1 = A  + (size_t)(m0 + 16 + l16) * K + half * 16;
    const __bf16* brow0 = BT + (size_t)(n0 + l16)      * K + half * 16;
    const __bf16* brow1 = BT + (size_t)(n0 + 16 + l16) * K + half * 16;

    v8f acc00 = {}, acc01 = {}, acc10 = {}, acc11 = {};
    for (int kb = 0; kb < K; kb += 32) {
        v16bf a0 = *(const v16bf*)(arow0 + kb);
        v16bf a1 = *(const v16bf*)(arow1 + kb);
        v16bf b0 = *(const v16bf*)(brow0 + kb);
        v16bf b1 = *(const v16bf*)(brow1 + kb);
        acc00 = __builtin_amdgcn_wmma_f32_16x16x32_bf16(false, a0, false, b0, (short)0, acc00, false, false);
        acc01 = __builtin_amdgcn_wmma_f32_16x16x32_bf16(false, a0, false, b1, (short)0, acc01, false, false);
        acc10 = __builtin_amdgcn_wmma_f32_16x16x32_bf16(false, a1, false, b0, (short)0, acc10, false, false);
        acc11 = __builtin_amdgcn_wmma_f32_16x16x32_bf16(false, a1, false, b1, (short)0, acc11, false, false);
    }

    auto store_tile = [&](const v8f& c, int mt, int nt) {
        int col = nt + l16;
        float bv = bias ? bias[col] : 0.0f;
        #pragma unroll
        for (int r = 0; r < 8; r++) {
            int row = mt + r + half * 8;     // C layout: lanes 16-31 hold M+8
            float v = c[r] + bv;
            if (relu) v = fmaxf(v, 0.0f);
            if (Cf)  Cf[(size_t)row * N + col]  = v;
            if (Cbf) Cbf[(size_t)row * N + col] = (__bf16)v;
        }
    };
    store_tile(acc00, m0,      n0);
    store_tile(acc01, m0,      n0 + 16);
    store_tile(acc10, m0 + 16, n0);
    store_tile(acc11, m0 + 16, n0 + 16);
}

// ---------------------------------------------------------------------------
// Attention, 1 block per pixel, wave = head. Keys = 25 gathered per-location
// rows; masked/OOB rows resolve to bias rows (projection of a zero row).
// masked != 0 : use act mask + score -1e9 (first attention)
// masked == 0 : no mask; OOB K rows come from kbias (second attention)
// ---------------------------------------------------------------------------
__global__ __launch_bounds__(256)
void attn_kernel(const float* __restrict__ Q, const float* __restrict__ K,
                 const float* __restrict__ V, const float* __restrict__ actf,
                 const float* __restrict__ kbias, const float* __restrict__ vbias,
                 __bf16* __restrict__ ctx, int masked)
{
    __shared__ float qsh[DIMD];
    __shared__ int nbr[32];
    __shared__ int msk[32];
    int pix = blockIdx.x;
    int i = pix >> 6, j = pix & 63;
    int t = threadIdx.x;
    qsh[t] = Q[(size_t)pix * DIMD + t];
    if (t < 32) {
        int np = -1, vld = 0;
        if (t < 25) {
            int ni = i + t / 5 - 2, nj = j + (t % 5) - 2;
            int inb = (ni >= 0 && ni < 64 && nj >= 0 && nj < 64);
            np  = inb ? (ni * 64 + nj) : -1;
            vld = inb ? (actf[inb ? (ni * 64 + nj) : 0] > 0.5f) : 0;
        }
        nbr[t] = np;
        msk[t] = vld;
    }
    __syncthreads();

    int h = t >> 5, lane = t & 31;
    const float scale = 0.1767766952966369f;    // 1/sqrt(32)
    float sc = -3.0e38f;                        // lanes >= 25 -> weight 0
    if (lane < 25) {
        if (masked && !msk[lane]) {
            sc = -1.0e9f;                       // reference mask value
        } else {
            const float* kr = (nbr[lane] >= 0) ? (K + (size_t)nbr[lane] * DIMD + h * 32)
                                               : (kbias + h * 32);
            float s = 0.f;
            #pragma unroll
            for (int d = 0; d < 32; d++) s += qsh[h * 32 + d] * kr[d];
            sc = s * scale;
        }
    }
    float mx = sc;
    #pragma unroll
    for (int o = 16; o > 0; o >>= 1) mx = fmaxf(mx, __shfl_xor(mx, o, 32));
    float e = __expf(sc - mx);
    float sum = e;
    #pragma unroll
    for (int o = 16; o > 0; o >>= 1) sum += __shfl_xor(sum, o, 32);
    float a = e / sum;

    float out = 0.f;
    for (int k = 0; k < 25; k++) {
        float ak = __shfl(a, k, 32);
        const float* vr = (nbr[k] >= 0) ? (V + (size_t)nbr[k] * DIMD) : vbias;
        out += ak * vr[h * 32 + lane];
    }
    ctx[(size_t)pix * DIMD + h * 32 + lane] = (__bf16)out;
}

// ---------------------------------------------------------------------------
// Residual add + LayerNorm (block = one row of 256). Optional bf16 mirror,
// optional activity gate (final layer) that also emits the act output.
// ---------------------------------------------------------------------------
__global__ __launch_bounds__(256)
void ln_kernel(const float* __restrict__ A, const float* __restrict__ B,
               const float* __restrict__ g, const float* __restrict__ b,
               float* __restrict__ outf, __bf16* __restrict__ outbf,
               const float* __restrict__ gate, float* __restrict__ gateout)
{
    __shared__ float red[8];
    int pix = blockIdx.x, t = threadIdx.x;
    size_t idx = (size_t)pix * DIMD + t;
    float v = A[idx] + B[idx];

    float s = v;
    #pragma unroll
    for (int o = 16; o > 0; o >>= 1) s += __shfl_xor(s, o, 32);
    if ((t & 31) == 0) red[t >> 5] = s;
    __syncthreads();
    float tot = 0.f;
    #pragma unroll
    for (int w = 0; w < 8; w++) tot += red[w];
    float mean = tot * (1.0f / 256.0f);
    float d = v - mean;
    float s2 = d * d;
    #pragma unroll
    for (int o = 16; o > 0; o >>= 1) s2 += __shfl_xor(s2, o, 32);
    __syncthreads();
    if ((t & 31) == 0) red[t >> 5] = s2;
    __syncthreads();
    float vtot = 0.f;
    #pragma unroll
    for (int w = 0; w < 8; w++) vtot += red[w];
    float var = vtot * (1.0f / 256.0f);

    float y = d * rsqrtf(var + 1e-5f) * g[t] + b[t];
    if (gate) {
        float gt = (gate[pix] > 0.5f) ? 1.0f : 0.0f;
        y *= gt;
        if (t == 0 && gateout) gateout[pix] = gt;
    }
    if (outf)  outf[idx]  = y;
    if (outbf) outbf[idx] = (__bf16)y;
}

// ---------------------------------------------------------------------------
extern "C" void kernel_launch(void* const* d_in, const int* in_sizes, int n_in,
                              void* d_out, int out_size, void* d_ws, size_t ws_size,
                              hipStream_t stream)
{
    (void)in_sizes; (void)n_in; (void)out_size; (void)ws_size;
    const float* x       = (const float*)d_in[0];
    const float* memory  = (const float*)d_in[1];
    const unsigned char* active = (const unsigned char*)d_in[2];
    const float* attn_w  = (const float*)d_in[3];
    const float* attn_b  = (const float*)d_in[4];   // [4][256]
    const float* pattn_w = (const float*)d_in[5];
    const float* pattn_b = (const float*)d_in[6];
    const float* ff_w1   = (const float*)d_in[7];
    const float* ff_b1   = (const float*)d_in[8];
    const float* ff_w2   = (const float*)d_in[9];
    const float* ff_b2   = (const float*)d_in[10];
    const float* ln_g    = (const float*)d_in[11];  // [3][256]
    const float* ln_b    = (const float*)d_in[12];

    char* base = (char*)d_ws;
    size_t off = 0;
    auto alloc = [&](size_t bytes) -> void* {
        size_t a = (off + 255) & ~(size_t)255;
        off = a + bytes;
        return (void*)(base + a);
    };
    const size_t ND = (size_t)NPIX * DIMD;
    float*  actf = (float*)alloc(NPIX * 4);
    float*  Xm   = (float*)alloc(ND * 4);
    __bf16* Xbf  = (__bf16*)alloc(ND * 2);
    __bf16* Mbf  = (__bf16*)alloc(ND * 2);
    __bf16* Wbf  = (__bf16*)alloc((size_t)(8 * 65536 + 2 * 131072) * 2);
    float*  XK   = (float*)alloc(ND * 4);
    float*  XV   = (float*)alloc(ND * 4);
    float*  Qb   = (float*)alloc(ND * 4);   // Q1, later Q2
    float*  MK   = (float*)alloc(ND * 4);
    float*  MV   = (float*)alloc(ND * 4);
    float*  S    = (float*)alloc(ND * 4);
    float*  S2   = (float*)alloc(ND * 4);
    float*  T    = (float*)alloc(ND * 4);
    __bf16* ctxbf= (__bf16*)alloc(ND * 2);
    __bf16* sbf  = (__bf16*)alloc(ND * 2);  // S bf16, later S2 bf16
    __bf16* Hbf  = (__bf16*)alloc((size_t)NPIX * 512 * 2);

    dim3 blk(256);
    auto gemm = [&](const __bf16* A, const __bf16* BT, const float* bias,
                    float* Cf, __bf16* Cbf, int N, int K, int relu) {
        dim3 grid(NPIX / 128, N / 64);
        gemm_bf16_kernel<<<grid, blk, 0, stream>>>(A, BT, bias, Cf, Cbf, N, K, relu);
    };

    // --- weight conversion / transposition (bf16) ---
    for (int w = 0; w < 4; w++) {
        wconv_kernel<<<256, blk, 0, stream>>>(attn_w  + (size_t)w * 65536, Wbf + (size_t)w       * 65536, 256, 256);
        wconv_kernel<<<256, blk, 0, stream>>>(pattn_w + (size_t)w * 65536, Wbf + (size_t)(4 + w) * 65536, 256, 256);
    }
    __bf16* WqT  = Wbf + 0 * 65536;
    __bf16* WkT  = Wbf + 1 * 65536;
    __bf16* WvT  = Wbf + 2 * 65536;
    __bf16* WoT  = Wbf + 3 * 65536;
    __bf16* Wq2T = Wbf + 4 * 65536;
    __bf16* Wk2T = Wbf + 5 * 65536;
    __bf16* Wv2T = Wbf + 6 * 65536;
    __bf16* Wo2T = Wbf + 7 * 65536;
    __bf16* W1T  = Wbf + 8 * 65536;            // [512][256]
    __bf16* W2T  = W1T + 131072;               // [256][512]
    wconv_kernel<<<512, blk, 0, stream>>>(ff_w1, W1T, 256, 512);
    wconv_kernel<<<512, blk, 0, stream>>>(ff_w2, W2T, 512, 256);

    // --- inputs ---
    prep_kernel<<<NPIX, blk, 0, stream>>>(x, memory, active, actf, Xm, Xbf, Mbf);

    // --- per-location projections (WMMA GEMMs) ---
    gemm(Xbf, WqT,  attn_b + 0,    Qb, nullptr, 256, 256, 0);   // Q1
    gemm(Xbf, WkT,  attn_b + 256,  XK, nullptr, 256, 256, 0);
    gemm(Xbf, WvT,  attn_b + 512,  XV, nullptr, 256, 256, 0);
    gemm(Mbf, Wk2T, pattn_b + 256, MK, nullptr, 256, 256, 0);
    gemm(Mbf, Wv2T, pattn_b + 512, MV, nullptr, 256, 256, 0);

    // --- attention 1 (masked, over x patches) + out proj + LN0 ---
    attn_kernel<<<NPIX, blk, 0, stream>>>(Qb, XK, XV, actf,
                                          attn_b + 256, attn_b + 512, ctxbf, 1);
    gemm(ctxbf, WoT, attn_b + 768, T, nullptr, 256, 256, 0);
    ln_kernel<<<NPIX, blk, 0, stream>>>(Xm, T, ln_g + 0, ln_b + 0, S, sbf, nullptr, nullptr);

    // --- attention 2 (unmasked, over memory patches) + out proj + LN2 ---
    gemm(sbf, Wq2T, pattn_b + 0, Qb, nullptr, 256, 256, 0);     // Q2
    attn_kernel<<<NPIX, blk, 0, stream>>>(Qb, MK, MV, actf,
                                          pattn_b + 256, pattn_b + 512, ctxbf, 0);
    gemm(ctxbf, Wo2T, pattn_b + 768, T, nullptr, 256, 256, 0);
    ln_kernel<<<NPIX, blk, 0, stream>>>(S, T, ln_g + 512, ln_b + 512, S2, sbf, nullptr, nullptr);

    // --- feed-forward (ReLU fused into FF1 store) + final gated LN1 ---
    gemm(sbf, W1T, ff_b1, nullptr, Hbf, 512, 256, 1);
    gemm(Hbf, W2T, ff_b2, T, nullptr, 256, 512, 0);

    float* outp = (float*)d_out;
    ln_kernel<<<NPIX, blk, 0, stream>>>(S2, T, ln_g + 256, ln_b + 256,
                                        outp, nullptr, actf, outp + ND);
}
